// RNNEncoder_76390288327237
// MI455X (gfx1250) — matrix operand order repaired and checked by reading, not compiled, not run on hardware
//
#include <hip/hip_runtime.h>
#include <hip/hip_bf16.h>
#include <math.h>

// Problem dims (fixed by reference)
#define B_   64
#define T_   2048
#define X_   128
#define H_   256
#define NG   4
#define KTOT (X_ + H_)     // 384 combined K (x then h)
#define NOUT (NG * H_)     // 1024 gate columns

// Persistent-kernel shape: 16 blocks x 4 waves (wave32) = 64 waves.
// Block b owns h-columns [16b, 16b+16); wave w owns batch rows [16w, 16w+16).
// Each wave computes all 4 gate tiles for its (batch-tile, h-tile) so the
// LSTM cell update stays in registers.
#define NBLK 16
#define TPB  128

typedef __attribute__((ext_vector_type(16))) __bf16 bf16x16;
typedef __attribute__((ext_vector_type(8)))  float  f32x8;

union Frag32B { uint4 q[2]; bf16x16 v; };   // 32 bytes = one WMMA A/B operand

__device__ __forceinline__ unsigned short f2bf(float f) {
  unsigned u = __builtin_bit_cast(unsigned, f);
  u += 0x7FFFu + ((u >> 16) & 1u);          // round-to-nearest-even
  return (unsigned short)(u >> 16);
}

__device__ __forceinline__ float sigmoidf_(float x) {
  return 1.0f / (1.0f + __expf(-x));
}
__device__ __forceinline__ float tanhf_(float x) {
  return 2.0f / (1.0f + __expf(-2.0f * x)) - 1.0f;
}

// ---------------------------------------------------------------------------
// One-shot pack: x -> bf16; Wt[n][k] = bf16( k<X ? Wx[k][n] : Wh[k-X][n] );
// zero h state; reset device barrier counter (harness replays w/o re-init).
// ---------------------------------------------------------------------------
__global__ void lstm_pack_kernel(const float* __restrict__ x,
                                 const float* __restrict__ Wx,
                                 const float* __restrict__ Wh,
                                 unsigned short* __restrict__ xbf,
                                 unsigned short* __restrict__ Wt,
                                 unsigned short* __restrict__ hbf,
                                 unsigned* __restrict__ cnt)
{
  const size_t stride = (size_t)gridDim.x * blockDim.x;
  const size_t i0 = (size_t)blockIdx.x * blockDim.x + threadIdx.x;
  if (i0 == 0) *cnt = 0u;

  const size_t NX = (size_t)B_ * T_ * X_;
  for (size_t i = i0; i < NX; i += stride) xbf[i] = f2bf(x[i]);

  const size_t NW = (size_t)NOUT * KTOT;
  for (size_t i = i0; i < NW; i += stride) {
    const size_t n = i / KTOT;
    const int    k = (int)(i - n * KTOT);
    const float  w = (k < X_) ? Wx[(size_t)k * NOUT + n]
                              : Wh[(size_t)(k - X_) * NOUT + n];
    Wt[i] = f2bf(w);
  }

  for (size_t i = i0; i < (size_t)B_ * H_; i += stride) hbf[i] = 0;
}

// ---------------------------------------------------------------------------
// Persistent LSTM kernel with per-timestep device-wide barrier.
// ---------------------------------------------------------------------------
__launch_bounds__(TPB, 1)
__global__ void lstm_persistent_kernel(const unsigned short* __restrict__ xbf, // [B][T][X] bf16
                                       const unsigned short* __restrict__ Wt,  // [NOUT][KTOT] bf16
                                       const float* __restrict__ bias,         // [NOUT]
                                       unsigned short* __restrict__ hbf,       // [B][H] bf16
                                       float* __restrict__ out,                // [B][T][H]
                                       unsigned* __restrict__ cnt)
{
  const int wave = threadIdx.x >> 5;
  const int lane = threadIdx.x & 31;
  const int lm   = lane & 15;     // column within tile (B-frag) / row (A-frag)
  const int lh   = lane >> 4;     // half-wave select
  const int h0   = blockIdx.x * 16;
  const int m0   = wave * 16;

  // Stage this block's weight columns into LDS, K-major per column:
  // ldsW[g][j][k] = Wt[g*H + h0 + j][k]   (4*16*384*2 = 48 KB)
  __shared__ __align__(16) unsigned short ldsW[NG][16][KTOT];
  for (int idx = threadIdx.x; idx < NG * 16; idx += TPB) {
    const int g = idx >> 4, j = idx & 15;
    const uint4* src = (const uint4*)(Wt + (size_t)(g * H_ + h0 + j) * KTOT);
    uint4* dst = (uint4*)&ldsW[g][j][0];
#pragma unroll
    for (int k = 0; k < KTOT / 8; ++k) dst[k] = src[k];
  }
  __syncthreads();

  float bi[NG];
#pragma unroll
  for (int g = 0; g < NG; ++g) bi[g] = bias[g * H_ + h0 + lm];

  const size_t xrow0 = (size_t)(m0 + lm) * T_ * X_;        // batch row in xbf
  const unsigned short* hrow = hbf + (size_t)(m0 + lm) * H_;

  f32x8 c = {};   // cell state, resident in VGPRs for the whole sequence

  for (int t = 0; t < T_; ++t) {
    f32x8 acc[NG] = {};

    const unsigned short* xrow = xbf + xrow0 + (size_t)t * X_;

    // ---- K over x-part (k = 0..127) ----
#pragma unroll
    for (int kk = 0; kk < X_ / 32; ++kk) {
      const int kb = kk * 32;
      Frag32B a;   // A(16x32): lane lm row, halves hold K {kb+8lh..} / {kb+16+8lh..}
      a.q[0] = *(const uint4*)&xrow[kb + lh * 8];
      a.q[1] = *(const uint4*)&xrow[kb + 16 + lh * 8];
#pragma unroll
      for (int g = 0; g < NG; ++g) {
        Frag32B b;   // B(32x16): lane lm col, halves hold K {kb+8lh..} / {kb+16+8lh..}
        const unsigned short* p = &ldsW[g][lm][kb + lh * 8];
        b.q[0] = *(const uint4*)p;
        b.q[1] = *(const uint4*)(p + 16);
        acc[g] = __builtin_amdgcn_wmma_f32_16x16x32_bf16(
            false, a.v, false, b.v, (short)0, acc[g], false, false);
      }
    }

    // ---- K over h-part (k = 128..383) ----
#pragma unroll
    for (int kk = 0; kk < H_ / 32; ++kk) {
      const int kb = kk * 32;
      Frag32B a;
      a.q[0] = *(const uint4*)&hrow[kb + lh * 8];
      a.q[1] = *(const uint4*)&hrow[kb + 16 + lh * 8];
#pragma unroll
      for (int g = 0; g < NG; ++g) {
        Frag32B b;
        const unsigned short* p = &ldsW[g][lm][X_ + kb + lh * 8];
        b.q[0] = *(const uint4*)p;
        b.q[1] = *(const uint4*)(p + 16);
        acc[g] = __builtin_amdgcn_wmma_f32_16x16x32_bf16(
            false, a.v, false, b.v, (short)0, acc[g], false, false);
      }
    }

    // ---- gate nonlinearities + cell/hidden update (in accumulator layout:
    //      element r -> batch m0 + r + 8*lh, column h0 + lm) ----
#pragma unroll
    for (int r = 0; r < 8; ++r) {
      const int batch = m0 + r + 8 * lh;
      const float ig = sigmoidf_(acc[0][r] + bi[0]);
      const float fg = sigmoidf_(acc[1][r] + bi[1]);
      const float gg = tanhf_  (acc[2][r] + bi[2]);
      const float og = sigmoidf_(acc[3][r] + bi[3]);
      const float cn = fg * c[r] + ig * gg;
      c[r] = cn;
      const float hn = og * tanhf_(cn);
      out[(size_t)batch * (T_ * H_) + (size_t)t * H_ + (h0 + lm)] = hn;
      hbf[(size_t)batch * H_ + (h0 + lm)] = f2bf(hn);
    }

    // ---- device-wide barrier: publish h, then wait for all 16 blocks ----
    __threadfence();                 // release: flush h stores to device scope
    __syncthreads();
    if (threadIdx.x == 0) {
      __atomic_fetch_add(cnt, 1u, __ATOMIC_RELEASE);
      const unsigned target = (unsigned)(t + 1) * (unsigned)NBLK;
      while (__atomic_load_n(cnt, __ATOMIC_ACQUIRE) < target) {
        __builtin_amdgcn_s_sleep(1);
      }
    }
    __syncthreads();
    __threadfence();                 // acquire: invalidate WGP-local caches
  }
}

// ---------------------------------------------------------------------------
// Launch: inputs are {x_seq, Wx, Wh, b}, all fp32.
// Workspace layout (≈34.5 MB): Wt bf16 | xbf bf16 | hbf bf16 | counter.
// ---------------------------------------------------------------------------
extern "C" void kernel_launch(void* const* d_in, const int* in_sizes, int n_in,
                              void* d_out, int out_size, void* d_ws, size_t ws_size,
                              hipStream_t stream) {
  const float* x  = (const float*)d_in[0];
  const float* Wx = (const float*)d_in[1];
  const float* Wh = (const float*)d_in[2];
  const float* b  = (const float*)d_in[3];
  float* out = (float*)d_out;

  char* ws = (char*)d_ws;
  size_t off = 0;
  auto carve = [&](size_t bytes) -> char* {
    char* p = ws + off;
    off = (off + bytes + 255) & ~(size_t)255;
    return p;
  };
  unsigned short* Wt  = (unsigned short*)carve((size_t)NOUT * KTOT * 2);       // 768 KB
  unsigned short* xbf = (unsigned short*)carve((size_t)B_ * T_ * X_ * 2);      // 32 MB
  unsigned short* hbf = (unsigned short*)carve((size_t)B_ * H_ * 2);           // 32 KB
  unsigned*       cnt = (unsigned*)carve(sizeof(unsigned));

  lstm_pack_kernel<<<1024, 256, 0, stream>>>(x, Wx, Wh, xbf, Wt, hbf, cnt);
  lstm_persistent_kernel<<<NBLK, TPB, 0, stream>>>(xbf, Wt, b, hbf, out, cnt);
}
